// PPGN_35210141893266
// MI455X (gfx1250) — matrix operand order repaired and checked by reading, not compiled, not run on hardware
//
#include <hip/hip_runtime.h>
#include <math.h>

// PPGN forward, fully fused: one workgroup (256 thr = 8 waves) per batch elem.
// Activations live in LDS as bf16 with padded channel stride (640) so that all
// GEMM epilogues are unconditional 16B ds_store_b128 (junk rows land in pad,
// masked off in registers at readout). All GEMMs: v_wmma_f32_16x16x32_bf16
// with f32 accumulate. Only inputs (~51MB) stream from HBM; output is 80KB.

#define NPIX 625      // 25*25 valid pixels
#define CST  640      // padded channel stride (elements); 640*2B = 16B-aligned
#define NMAX 25
#define NCH  32
#define BLK  256

typedef __attribute__((ext_vector_type(16))) __bf16        v16bf;
typedef __attribute__((ext_vector_type(8)))  float         v8f;
typedef __attribute__((ext_vector_type(4)))  float         v4f;
typedef __attribute__((ext_vector_type(4)))  unsigned int  v4u;

union ABu { v16bf v; unsigned short u[16]; unsigned int w[8]; };

__device__ __forceinline__ unsigned int rnd_bf(float f) {   // RNE, bf16 in [31:16]
  unsigned int u = __float_as_uint(f);
  return u + 0x7FFFu + ((u >> 16) & 1u);
}
__device__ __forceinline__ unsigned short f2bf(float f) {
  return (unsigned short)(rnd_bf(f) >> 16);
}
__device__ __forceinline__ float bf2f(unsigned short h) {
  return __uint_as_float(((unsigned int)h) << 16);
}
__device__ __forceinline__ unsigned int pack_bf(float lo, float hi) {
  return (rnd_bf(lo) >> 16) | (rnd_bf(hi) & 0xFFFF0000u);
}

__device__ __forceinline__ v8f wmma_bf16(const ABu& a, const ABu& b, v8f c) {
  return __builtin_amdgcn_wmma_f32_16x16x32_bf16(
      false, a.v, false, b.v, (short)0, c, false, false);
}

// GEMM: OUT[n][p] = relu((sum_k A[k][p] * w[n][k] + bias[n]) * msum[p])
// A = concat(A0 [C0 ch], A1 [C1 ch]), channel-major [ch][CST] bf16 in LDS.
// K zero-padded in registers; M rows >= 625 spill into the per-channel pad.
template<int C0, int C1>
__device__ __forceinline__ void conv_gemm(
    const unsigned short* A0, const unsigned short* A1,
    const float* __restrict__ gw, const float* __restrict__ gb,
    unsigned short* OUT, unsigned short* sWh, float* sB, const float* sMsum)
{
  constexpr int Kt = C0 + C1;
  constexpr int KS = (Kt + 31) / 32;
  const int tid = threadIdx.x;
  for (int i = tid; i < NCH * Kt; i += BLK) sWh[i] = f2bf(gw[i]);
  if (tid < NCH) sB[tid] = gb[tid];
  __syncthreads();

  const int lane = tid & 31, wave = tid >> 5;
  const int hl = lane >> 4, l16 = lane & 15;
  const int dA = (int)(A1 - A0);     // both in LDS; C1==0 passes A1==A0

  for (int job = wave; job < 80; job += 8) {   // 40 M-tiles x 2 N-tiles, uniform
    const int mt = job >> 1, nt = job & 1;
    int p = mt * 16 + l16; if (p > NPIX - 1) p = NPIX - 1;  // A reads stay valid
    const int n = nt * 16 + l16;
    const float bias = sB[n];
    const unsigned short* wrow = sWh + n * Kt;
    v8f acc = {0.f, 0.f, 0.f, 0.f, 0.f, 0.f, 0.f, 0.f};
#pragma unroll
    for (int ks = 0; ks < KS; ++ks) {
      ABu a, b;
#pragma unroll
      for (int q = 0; q < 16; ++q) {           // branchless clamped gathers
        const int v = q >> 1, e = q & 1;
        const int ka = ks * 32 + ((v & 4) ? 16 : 0) + 2 * (v & 3) + 8 * hl + e;
        const int kc = (ka < Kt) ? ka : 0;
        const int off = (C1 == 0 || kc < C0) ? kc * CST : (dA + (kc - C0) * CST);
        const unsigned short av = A0[off + p];
        a.u[q] = (ka < Kt) ? av : (unsigned short)0;
      }
#pragma unroll
      for (int v = 0; v < 8; ++v) {            // bf16 K-pairs: aligned b32 loads
        const int kbB = ks * 32 + ((v & 4) ? 16 : 0) + 2 * (v & 3) + 8 * hl;
        const int kbc = (kbB < Kt) ? kbB : 0;
        const unsigned int wv = *(const unsigned int*)(wrow + kbc);
        b.w[v] = (kbB < Kt) ? wv : 0u;
      }
      acc = wmma_bf16(a, b, acc);
    }
    // Uniform epilogue: rows >= 625 fall in [625,640) pad, masked at readout.
    const int pd0 = mt * 16 + 8 * hl;
    const v4f ms0 = *(const v4f*)(sMsum + pd0);
    const v4f ms1 = *(const v4f*)(sMsum + pd0 + 4);
    float r[8];
#pragma unroll
    for (int v = 0; v < 4; ++v) r[v]     = fmaxf((acc[v]     + bias) * ms0[v], 0.f);
#pragma unroll
    for (int v = 0; v < 4; ++v) r[4 + v] = fmaxf((acc[4 + v] + bias) * ms1[v], 0.f);
    v4u pk;
    pk.x = pack_bf(r[0], r[1]); pk.y = pack_bf(r[2], r[3]);
    pk.z = pack_bf(r[4], r[5]); pk.w = pack_bf(r[6], r[7]);
    *(v4u*)(OUT + n * CST + pd0) = pk;         // one ds_store_b128
  }
  __syncthreads();
}

// Per-channel 25x25x25 matmul: T1[c] = (T1[c] @ T2[c]) * msum, in place.
__device__ __forceinline__ void spatial_matmul(
    unsigned short* T1, const unsigned short* T2, const float* sMsum)
{
  const int tid = threadIdx.x, lane = tid & 31, wave = tid >> 5;
  const int hl = lane >> 4, l16 = lane & 15;
  const int i1 = (16 + l16 < NMAX) ? 16 + l16 : NMAX - 1;   // clamped row, mt=1

  for (int c = wave; c < NCH; c += 8) {        // 4 channels per wave, uniform
    const unsigned short* A = T1 + c * CST;
    const unsigned short* B = T2 + c * CST;
    ABu a0, a1, b0, b1;
#pragma unroll
    for (int q = 0; q < 16; ++q) {
      const int v = q >> 1, e = q & 1;
      const int k = ((v & 4) ? 16 : 0) + 2 * (v & 3) + 8 * hl + e;
      const int kc = (k < NMAX) ? k : 0;
      const unsigned short x0 = A[l16 * NMAX + kc];
      const unsigned short x1 = A[i1  * NMAX + kc];
      a0.u[q] = (k < NMAX) ? x0 : (unsigned short)0;
      a1.u[q] = (k < NMAX) ? x1 : (unsigned short)0;
      const int kB = ((v & 4) ? 16 : 0) + 2 * (v & 3) + 8 * hl + e;
      const int kBc = (kB < NMAX) ? kB : 0;
      const int j1 = (16 + l16 < NMAX) ? 16 + l16 : NMAX - 1;
      const unsigned short y0 = B[kBc * NMAX + l16];
      const unsigned short y1 = B[kBc * NMAX + j1];
      b0.u[q] = (kB < NMAX) ? y0 : (unsigned short)0;
      b1.u[q] = (kB < NMAX) ? y1 : (unsigned short)0;
    }
    const v8f zz = {0.f, 0.f, 0.f, 0.f, 0.f, 0.f, 0.f, 0.f};
    v8f d[4];
    d[0] = wmma_bf16(a0, b0, zz);
    d[1] = wmma_bf16(a0, b1, zz);
    d[2] = wmma_bf16(a1, b0, zz);
    d[3] = wmma_bf16(a1, b1, zz);

    unsigned short* W = T1 + c * CST;          // frags already in registers
#pragma unroll
    for (int mt = 0; mt < 2; ++mt)
#pragma unroll
      for (int nt = 0; nt < 2; ++nt)
#pragma unroll
        for (int v = 0; v < 8; ++v) {          // branchless: invalid -> pad slot
          const int i = mt * 16 + v + 8 * hl;
          const int j = nt * 16 + l16;
          const bool ok = (i < NMAX) && (j < NMAX);
          const int si = ok ? (i * NMAX + j) : 632;
          W[si] = f2bf(d[mt * 2 + nt][v] * sMsum[si]);
        }
  }
  __syncthreads();
}

// Masked readout: sS[blk*64 + k*32 + c] += sum_p X[c][p] * m[k][p]
__device__ __forceinline__ void readout(const unsigned short* X, int blk,
                                        const float* sM, float* sS)
{
  const int tid = threadIdx.x, lane = tid & 31;
  const bool act = tid < 157;                  // 157 groups of 4 pixels
  const int p0 = 4 * (act ? tid : 156);
#pragma unroll 1
  for (int c = 0; c < NCH; ++c) {
    const unsigned long long xl = *(const unsigned long long*)(X + c * CST + p0);
    const v4f m0 = *(const v4f*)(sM + p0);
    const v4f m1 = *(const v4f*)(sM + CST + p0);
    float a0 = 0.f, a1 = 0.f;
#pragma unroll
    for (int e = 0; e < 4; ++e) {
      const float xv = bf2f((unsigned short)((xl >> (16 * e)) & 0xFFFFu));
      const bool ok = act && (p0 + e < NPIX);
      a0 += ok ? xv * m0[e] : 0.f;
      a1 += ok ? xv * m1[e] : 0.f;
    }
#pragma unroll
    for (int off = 16; off > 0; off >>= 1) {
      a0 += __shfl_down(a0, off, 32);
      a1 += __shfl_down(a1, off, 32);
    }
    if (lane == 0) {
      atomicAdd(&sS[blk * 64 + c], a0);        // ds_add_f32
      atomicAdd(&sS[blk * 64 + 32 + c], a1);
    }
  }
  __syncthreads();
}

template<int CIN>
__device__ __forceinline__ void ppgn_block(
    const unsigned short* IN,
    unsigned short* T1, unsigned short* T2, unsigned short* OUT,
    const float* wa, const float* ba, const float* wb, const float* bb,
    const float* wc, const float* bc,
    unsigned short* sWh, float* sB, const float* sMsum)
{
  conv_gemm<CIN, 0>(IN, IN, wa, ba, T1, sWh, sB, sMsum);       // x1
  conv_gemm<CIN, 0>(IN, IN, wb, bb, T2, sWh, sB, sMsum);       // x2
  spatial_matmul(T1, T2, sMsum);                               // x12 (in T1)
  conv_gemm<NCH, CIN>(T1, IN, wc, bc, OUT, sWh, sB, sMsum);    // concat conv
}

__global__ __launch_bounds__(BLK) void ppgn_kernel(
    const float* __restrict__ gx, const float* __restrict__ gm,
    const float* w11, const float* b11, const float* w12, const float* b12,
    const float* w13, const float* b13,
    const float* w21, const float* b21, const float* w22, const float* b22,
    const float* w23, const float* b23,
    const float* w31, const float* b31, const float* w32, const float* b32,
    const float* w33, const float* b33,
    const float* wh, const float* bh, float* __restrict__ out)
{
  __shared__ __align__(16) unsigned short sX[NCH * CST];   // 40 KB each
  __shared__ __align__(16) unsigned short sY[NCH * CST];
  __shared__ __align__(16) unsigned short sZ[NCH * CST];
  __shared__ __align__(16) float sM[2 * CST];
  __shared__ __align__(16) float sMsum[CST];
  __shared__ __align__(16) unsigned short sWh[NCH * 64];
  __shared__ __align__(16) float sB[NCH];
  __shared__ __align__(16) float sS[192];

  const int tid = threadIdx.x;
  const int b = blockIdx.x;

  for (int p = tid; p < NPIX; p += BLK) {
    const float m0 = gm[(size_t)b * 2 * NPIX + p];
    const float m1 = gm[(size_t)b * 2 * NPIX + NPIX + p];
    sM[p] = m0; sM[CST + p] = m1; sMsum[p] = m0 + m1;
  }
  for (int t = tid; t < (8 * NPIX) / 4; t += BLK) {        // global_load_b128
    const v4f xv = *(const v4f*)(gx + (size_t)b * 8 * NPIX + 4 * t);
#pragma unroll
    for (int e = 0; e < 4; ++e) {
      const int idx = 4 * t + e;
      const int c = idx / NPIX;
      sX[c * CST + (idx - c * NPIX)] = f2bf(xv[e]);
    }
  }
  if (tid < 192) sS[tid] = 0.f;
  __syncthreads();

  // buffer rotation: IN -> (T1,T2) -> OUT reuses T2's storage
  ppgn_block<8> (sX, sY, sZ, sZ, w11, b11, w12, b12, w13, b13, sWh, sB, sMsum);
  readout(sZ, 0, sM, sS);
  ppgn_block<32>(sZ, sX, sY, sY, w21, b21, w22, b22, w23, b23, sWh, sB, sMsum);
  readout(sY, 1, sM, sS);
  ppgn_block<32>(sY, sZ, sX, sX, w31, b31, w32, b32, w33, b33, sWh, sB, sMsum);
  readout(sX, 2, sM, sS);

  if (tid < 10) {
    float acc = bh[tid];
#pragma unroll 1
    for (int r = 0; r < 192; ++r) acc += sS[r] * wh[tid * 192 + r];
    out[(size_t)b * 10 + tid] = tanhf(acc);
  }
}

extern "C" void kernel_launch(void* const* d_in, const int* in_sizes, int n_in,
                              void* d_out, int out_size, void* d_ws, size_t ws_size,
                              hipStream_t stream) {
  (void)in_sizes; (void)n_in; (void)out_size; (void)d_ws; (void)ws_size;
  const float* gx  = (const float*)d_in[0];
  const float* gm  = (const float*)d_in[1];
  const float* w11 = (const float*)d_in[2];  const float* b11 = (const float*)d_in[3];
  const float* w12 = (const float*)d_in[4];  const float* b12 = (const float*)d_in[5];
  const float* w13 = (const float*)d_in[6];  const float* b13 = (const float*)d_in[7];
  const float* w21 = (const float*)d_in[8];  const float* b21 = (const float*)d_in[9];
  const float* w22 = (const float*)d_in[10]; const float* b22 = (const float*)d_in[11];
  const float* w23 = (const float*)d_in[12]; const float* b23 = (const float*)d_in[13];
  const float* w31 = (const float*)d_in[14]; const float* b31 = (const float*)d_in[15];
  const float* w32 = (const float*)d_in[16]; const float* b32 = (const float*)d_in[17];
  const float* w33 = (const float*)d_in[18]; const float* b33 = (const float*)d_in[19];
  const float* wh  = (const float*)d_in[20]; const float* bh  = (const float*)d_in[21];

  ppgn_kernel<<<2048, BLK, 0, stream>>>(
      gx, gm, w11, b11, w12, b12, w13, b13,
      w21, b21, w22, b22, w23, b23,
      w31, b31, w32, b32, w33, b33,
      wh, bh, (float*)d_out);
}
